// MambaBlock_11725260718322
// MI455X (gfx1250) — matrix operand order repaired and checked by reading, not compiled, not run on hardware
//
#include <hip/hip_runtime.h>
#include <hip/hip_bf16.h>
#include <stdint.h>

typedef __attribute__((ext_vector_type(16))) __bf16 v16bf;
typedef __attribute__((ext_vector_type(8)))  float  v8f;
typedef __attribute__((ext_vector_type(4)))  unsigned int v4u;
typedef __attribute__((ext_vector_type(8)))  int v8i;
typedef __attribute__((ext_vector_type(4)))  int v4i;

#define D_MODEL 1024
#define D_INNER 2048
#define D_STATE 64
#define SEQ_L   512
#define BATCH   2
#define ROWS    (BATCH*SEQ_L)   /* 1024 */
#define XP_LD   132             /* padded row stride for xp (129 cols) */

#if defined(__HIP_DEVICE_COMPILE__) && \
    __has_builtin(__builtin_amdgcn_tensor_load_to_lds) && \
    __has_builtin(__builtin_amdgcn_s_wait_tensorcnt)
#define USE_TDM 1
#pragma message("gfx1250: TDM tensor_load_to_lds path ENABLED (device pass)")
#else
#define USE_TDM 0
#endif

#if USE_TDM
#define NBUF 2
#else
#define NBUF 1
#endif

__device__ __forceinline__ float sigmoidf_(float x) { return 1.0f / (1.0f + __expf(-x)); }

__device__ __forceinline__ uint32_t lds_addr_of(const void* p) {
#if defined(__HIP_DEVICE_COMPILE__)
    return (uint32_t)(uintptr_t)(__attribute__((address_space(3))) const void*)p;
#else
    return 0;
#endif
}

// TDM 2D tile load: `rows` x `tile_cols` f32 tile, row stride `stride` (floats),
// from global `gsrc` into LDS byte offset `lds_off` (packed rows x tile_cols).
// Columns >= valid_cols are zero-filled by the TDM OOB rule (tensor_dim0).
__device__ __forceinline__ void tdm_load_2d(const float* gsrc, uint32_t lds_off,
                                            int tile_cols, int valid_cols,
                                            int rows, int stride)
{
#if USE_TDM
    uint64_t ga = (uint64_t)(uintptr_t)gsrc;
    // D# group0: count=1 | lds_addr[63:32] | global_addr[120:64] | type=2[127:126]
    v4u g0 = { 1u,
               lds_off,
               (unsigned)(ga & 0xFFFFFFFFu),
               (unsigned)(((ga >> 32) & 0x1FFFFFFu) | (2u << 30)) };
    // D# group1
    v8i g1 = { (int)(2u << 16),                   // [17:16] data_size = 2 -> 4B elems
               (int)((unsigned)valid_cols << 16), // [79:48] tensor_dim0 (low 16)
               (int)((unsigned)rows << 16),       // [111:80] tensor_dim1 (low 16)
               (int)((unsigned)tile_cols << 16),  // [127:112] tile_dim0
               (int)rows,                         // [143:128] tile_dim1 ([159:144] tile_dim2=0)
               (int)stride,                       // [207:160] tensor_dim0_stride (low 32)
               0, 0 };
    v4i g2 = {0, 0, 0, 0};
    v4i g3 = {0, 0, 0, 0};
#if __clang_major__ >= 23
    v8i g4 = {0, 0, 0, 0, 0, 0, 0, 0};
    __builtin_amdgcn_tensor_load_to_lds(g0, g1, g2, g3, g4, 0);
#else
    __builtin_amdgcn_tensor_load_to_lds(g0, g1, g2, g3, 0);
#endif
#else
    (void)gsrc; (void)lds_off; (void)tile_cols; (void)valid_cols; (void)rows; (void)stride;
#endif
}

// Issue both tiles for K-step `step` into ping-pong buffer (step & 1).
__device__ __forceinline__ void issue_tiles(const float* A, const float* B,
                                            int m0, int n0, int N, int lda, int ldb,
                                            int step, uint32_t a_lds, uint32_t b_lds)
{
#if USE_TDM
    const int k0  = step << 5;
    const int buf = step & 1;
    // A tile: 64 rows x 32 cols
    tdm_load_2d(A + (size_t)m0 * lda + k0, a_lds + (unsigned)buf * 8192u, 32, 32, 64, lda);
    // B tile: 32 rows x 64 cols, zero-fill cols >= N
    int vb = N - n0; if (vb > 64) vb = 64;
    tdm_load_2d(B + (size_t)k0 * ldb + n0, b_lds + (unsigned)buf * 8192u, 64, vb, 32, ldb);
#else
    (void)A; (void)B; (void)m0; (void)n0; (void)N; (void)lda; (void)ldb;
    (void)step; (void)a_lds; (void)b_lds;
#endif
}

// C[M,N] = A[M,K] @ B[K,N], fp32 in/out, bf16 WMMA compute.
// Block: 128 threads (4 waves). Block tile: 64(M) x 64(N). K step: 32.
// Wave w computes rows [w*16, w*16+16) x all 64 cols -> 4 WMMAs per K step.
// TDM path: double-buffered LDS, DMA of step i+1 overlaps compute of step i;
// last iteration peeled so the hot loop is branch-free.
__global__ void __launch_bounds__(128)
gemm_bf16_wmma(const float* __restrict__ A, const float* __restrict__ B,
               float* __restrict__ C, int M, int N, int K,
               int lda, int ldb, int ldc)
{
    __shared__ float Asf[NBUF][64][32];   // [m][k] f32
    __shared__ float Bsf[NBUF][32][64];   // [k][n] f32

    const int tid  = threadIdx.x;
    const int lane = tid & 31;
    const int wave = tid >> 5;        // 0..3
    const int half = lane >> 4;       // 0 or 1
    const int lrow = lane & 15;

    const int m0 = blockIdx.x * 64;
    const int n0 = blockIdx.y * 64;
    const int arow = wave * 16 + lrow;

    v8f acc0 = {0.f,0.f,0.f,0.f,0.f,0.f,0.f,0.f};
    v8f acc1 = acc0, acc2 = acc0, acc3 = acc0;

    const int nsteps = K >> 5;

    // Fragment build + 4 WMMAs for buffer `cur`.
    auto compute_step = [&](int cur) {
        // A fragment (16x32 bf16 layout): lanes 0-15 -> K {0..7,16..23},
        // lanes 16-31 -> K {8..15,24..31}; convert f32->bf16 in regs.
        v16bf af;
        #pragma unroll
        for (int t = 0; t < 8; ++t) {
            af[t]     = (__bf16)Asf[cur][arow][half * 8 + t];
            af[8 + t] = (__bf16)Asf[cur][arow][half * 8 + 16 + t];
        }
        // B fragments: col = lane&15; lanes 0-15 K {0..7,16..23}, lanes 16-31 K {8..15,24..31}
        v16bf bf0, bf1, bf2, bf3;
        #pragma unroll
        for (int t = 0; t < 16; ++t) {
            int kq = half * 8 + (t & 7) + (t >> 3) * 16;
            bf0[t] = (__bf16)Bsf[cur][kq][ 0 + lrow];
            bf1[t] = (__bf16)Bsf[cur][kq][16 + lrow];
            bf2[t] = (__bf16)Bsf[cur][kq][32 + lrow];
            bf3[t] = (__bf16)Bsf[cur][kq][48 + lrow];
        }
        acc0 = __builtin_amdgcn_wmma_f32_16x16x32_bf16(false, af, false, bf0, (short)0, acc0, false, false);
        acc1 = __builtin_amdgcn_wmma_f32_16x16x32_bf16(false, af, false, bf1, (short)0, acc1, false, false);
        acc2 = __builtin_amdgcn_wmma_f32_16x16x32_bf16(false, af, false, bf2, (short)0, acc2, false, false);
        acc3 = __builtin_amdgcn_wmma_f32_16x16x32_bf16(false, af, false, bf3, (short)0, acc3, false, false);
    };

#if USE_TDM
    const uint32_t a_lds = lds_addr_of(&Asf[0][0][0]);
    const uint32_t b_lds = lds_addr_of(&Bsf[0][0][0]);
    if (wave == 0)
        issue_tiles(A, B, m0, n0, N, lda, ldb, 0, a_lds, b_lds);

    // Steady state: unconditionally prefetch step i+1, wait for step i's pair.
    for (int i = 0; i < nsteps - 1; ++i) {
        if (wave == 0) {
            issue_tiles(A, B, m0, n0, N, lda, ldb, i + 1, a_lds, b_lds);
            __builtin_amdgcn_s_wait_tensorcnt((short)2);   // step i's pair landed
        }
        __syncthreads();
        compute_step(i & 1);
        __syncthreads();
    }
    // Epilogue: last K-step.
    if (wave == 0)
        __builtin_amdgcn_s_wait_tensorcnt((short)0);
    __syncthreads();
    compute_step((nsteps - 1) & 1);
#else
    for (int i = 0; i < nsteps; ++i) {
        const int k0 = i << 5;
        #pragma unroll
        for (int j = 0; j < 16; ++j) {
            int e = tid + 128 * j;          // 0..2047
            int r = e >> 5, kk = e & 31;
            Asf[0][r][kk] = A[(size_t)(m0 + r) * lda + k0 + kk];
        }
        #pragma unroll
        for (int j = 0; j < 16; ++j) {
            int e  = tid + 128 * j;
            int kk = e >> 6, nn = e & 63;
            int ng = n0 + nn;
            int nc = (ng < N) ? ng : (N - 1);          // clamp: branchless, safe addr
            float v = B[(size_t)(k0 + kk) * ldb + nc];
            Bsf[0][kk][nn] = (ng < N) ? v : 0.0f;
        }
        __syncthreads();
        compute_step(0);
        __syncthreads();
    }
#endif

    // C/D layout: VGPR r -> row r + 8*half, col = lane&15
    #pragma unroll
    for (int r = 0; r < 8; ++r) {
        int row = m0 + wave * 16 + r + 8 * half;
        float* Crow = C + (size_t)row * ldc;
        int c0 = n0 + lrow;
        if (c0      < N) Crow[c0]      = acc0[r];
        if (c0 + 16 < N) Crow[c0 + 16] = acc1[r];
        if (c0 + 32 < N) Crow[c0 + 32] = acc2[r];
        if (c0 + 48 < N) Crow[c0 + 48] = acc3[r];
    }
}

// Depthwise causal conv (K=4) + bias + SiLU on xs half of xz.
__global__ void conv_silu_kernel(const float* __restrict__ xz,
                                 const float* __restrict__ cw,
                                 const float* __restrict__ cb,
                                 float* __restrict__ xs)
{
    int idx = blockIdx.x * blockDim.x + threadIdx.x;
    if (idx >= ROWS * D_INNER) return;
    int d  = idx & (D_INNER - 1);
    int rl = idx >> 11;              // b*512 + l
    int l  = rl & (SEQ_L - 1);
    float acc = cb[d];
    #pragma unroll
    for (int k = 0; k < 4; ++k) {
        int ls = l - 3 + k;
        if (ls >= 0)
            acc += cw[d * 4 + k] * xz[(size_t)(rl - l + ls) * (2 * D_INNER) + d];
    }
    xs[idx] = acc * sigmoidf_(acc);
}

// Selective scan: one wave32 per (b, d) channel; each lane owns 2 of 64 states.
// Fuses dt=softplus(dt_raw*dt_w+dt_b), D-skip, and SiLU(z) gating.
__global__ void __launch_bounds__(256)
scan_kernel(const float* __restrict__ xz, const float* __restrict__ xs,
            const float* __restrict__ xp, const float* __restrict__ dt_w,
            const float* __restrict__ dt_b, const float* __restrict__ A_log,
            const float* __restrict__ Dp, float* __restrict__ y,
            float* __restrict__ state_out)
{
    const int lane = threadIdx.x & 31;
    const int wid  = blockIdx.x * (blockDim.x >> 5) + (threadIdx.x >> 5); // 0..4095
    const int b  = wid >> 11;
    const int d  = wid & (D_INNER - 1);
    const int n0 = lane * 2;

    float A0 = -__expf(A_log[d * D_STATE + n0]);
    float A1 = -__expf(A_log[d * D_STATE + n0 + 1]);
    float w_dt = dt_w[d], b_dt = dt_b[d], Dd = Dp[d];
    float s0 = 0.f, s1 = 0.f;

    for (int l = 0; l < SEQ_L; ++l) {
        int rl = b * SEQ_L + l;
        float xt  = xs[(size_t)rl * D_INNER + d];
        float dtr = xp[(size_t)rl * XP_LD + 128];
        float dt  = dtr * w_dt + b_dt;
        dt = (dt > 20.f) ? dt : log1pf(__expf(dt));       // softplus
        float2 Bn = *(const float2*)(xp + (size_t)rl * XP_LD + n0);
        float2 Cn = *(const float2*)(xp + (size_t)rl * XP_LD + D_STATE + n0);
        float dx = dt * xt;
        s0 = __expf(dt * A0) * s0 + dx * Bn.x;
        s1 = __expf(dt * A1) * s1 + dx * Bn.y;
        float part = s0 * Cn.x + s1 * Cn.y;
        #pragma unroll
        for (int off = 16; off; off >>= 1)
            part += __shfl_xor(part, off, 32);
        if (lane == 0) {
            float zv = xz[(size_t)rl * (2 * D_INNER) + D_INNER + d];
            y[(size_t)rl * D_INNER + d] = (part + Dd * xt) * (zv * sigmoidf_(zv));
        }
    }
    state_out[((size_t)b * D_INNER + d) * D_STATE + n0]     = s0;
    state_out[((size_t)b * D_INNER + d) * D_STATE + n0 + 1] = s1;
}

extern "C" void kernel_launch(void* const* d_in, const int* in_sizes, int n_in,
                              void* d_out, int out_size, void* d_ws, size_t ws_size,
                              hipStream_t stream)
{
    const float* x       = (const float*)d_in[0];
    const float* in_w    = (const float*)d_in[1];
    const float* conv_w  = (const float*)d_in[2];
    const float* conv_b  = (const float*)d_in[3];
    const float* xproj_w = (const float*)d_in[4];
    const float* dt_w    = (const float*)d_in[5];
    const float* dt_b    = (const float*)d_in[6];
    const float* A_log   = (const float*)d_in[7];
    const float* Dp      = (const float*)d_in[8];
    const float* out_w   = (const float*)d_in[9];

    float* ws = (float*)d_ws;
    float* xz = ws;                                  // 1024*4096
    float* xs = xz + (size_t)ROWS * 2 * D_INNER;     // 1024*2048
    float* xp = xs + (size_t)ROWS * D_INNER;         // 1024*132
    float* y  = xp + (size_t)ROWS * XP_LD;           // 1024*2048

    float* out   = (float*)d_out;                    // (B,L,d_model)
    float* stout = out + (size_t)ROWS * D_MODEL;     // (B,d_inner,d_state)

    // 1) xz = x @ in_proj_w : (1024,1024)x(1024,4096)
    gemm_bf16_wmma<<<dim3(ROWS / 64, (2 * D_INNER) / 64), 128, 0, stream>>>(
        x, in_w, xz, ROWS, 2 * D_INNER, D_MODEL, D_MODEL, 2 * D_INNER, 2 * D_INNER);

    // 2) causal depthwise conv + SiLU
    conv_silu_kernel<<<(ROWS * D_INNER) / 256, 256, 0, stream>>>(xz, conv_w, conv_b, xs);

    // 3) xp = xs @ x_proj_w : (1024,2048)x(2048,129), padded ldc
    gemm_bf16_wmma<<<dim3(ROWS / 64, (129 + 63) / 64), 128, 0, stream>>>(
        xs, xproj_w, xp, ROWS, 129, D_INNER, D_INNER, 129, XP_LD);

    // 4) selective scan + gating (also writes final state output)
    scan_kernel<<<(BATCH * D_INNER) / 8, 256, 0, stream>>>(
        xz, xs, xp, dt_w, dt_b, A_log, Dp, y, stout);

    // 5) out = y @ out_proj_w : (1024,2048)x(2048,1024)
    gemm_bf16_wmma<<<dim3(ROWS / 64, D_MODEL / 64), 128, 0, stream>>>(
        y, out_w, out, ROWS, D_MODEL, D_INNER, D_INNER, D_MODEL, D_MODEL);
}
